// WassersteinPersistentHomologyLoss_74483322847893
// MI455X (gfx1250) — matrix operand order repaired and checked by reading, not compiled, not run on hardware
//
#include <hip/hip_runtime.h>

// ---------------------------------------------------------------------------
// WassersteinPersistentHomologyLoss for MI455X (gfx1250, wave32, WMMA).
//
// Pipeline:
//   1) pool_kernel      : softmax(x) + one-hot(y), 8x8 avg-pool -> mx,my [12][32][32]
//   2) diagram_kernel   : 48 runs of H0 persistence (sublevel/superlevel, x/y),
//                         bitonic sort + serial Kruskal + top-256 by persistence
//   3) eps_kernel       : eps_p = 0.02*max(max(C_p),1e-6)  (24 problems)
//   4) buildK_kernel    : K = exp(-C/eps) in bf16, row-major + transposed copy
//   5) sinkhorn_kernel  : 150 iterations of u=1/(Kv), v=1/(K^T u) using
//                         v_wmma_f32_16x16x32_bf16 (vector replicated over N=16),
//                         then sum(u_i K_ij C_ij v_j) -> partial[p]
//   6) final_kernel     : out = sum(partial)/B
//
// Workspace layout (bytes):  requires ~25.5 MB
//   mx      @ 0         12*1024*4      = 49152
//   my      @ 49152                    = 49152
//   diag    @ 98304     24*4*256*4     = 98304   ([p][ab|ad|bb|bd][256])
//   eps     @ 196608    24*4
//   partial @ 196864    24*4
//   Kf      @ 262144    24*512*512*2   = 12582912  (bf16 bits)
//   Kt      @ 12845056  24*512*512*2   = 12582912
// ---------------------------------------------------------------------------

typedef __attribute__((ext_vector_type(16))) __bf16    v16bf;
typedef __attribute__((ext_vector_type(8)))  float     v8f;
typedef __attribute__((ext_vector_type(4)))  unsigned  u32x4;   // builtin vector,
                                                                // AS-qualified loads OK

struct U4x2 { u32x4 lo, hi; };
static_assert(sizeof(U4x2) == 32, "bitcast size");

// explicit address-space pointers (avoid FLAT path / null-check overhead in
// the __noinline__ matvec; LDS generic addresses truncate to 32-bit offsets)
typedef const u32x4 __attribute__((address_space(1)))* GQ4;   // global
typedef const u32x4 __attribute__((address_space(3)))* LQ4;   // LDS read
typedef float       __attribute__((address_space(3)))* LF;    // LDS write

#define NPIX  1024
#define NE    1984
#define SORTN 2048
#define KTOP  256
#define TWOK  512
#define NPROB 24
#define SINK_ITERS 150

// ---- bf16 helpers (storage type = unsigned short, RNE) ---------------------
__device__ __forceinline__ unsigned short f2bf(float f) {
  unsigned u = __float_as_uint(f);
  unsigned r = (u + 0x7FFFu + ((u >> 16) & 1u)) >> 16;
  return (unsigned short)r;
}
__device__ __forceinline__ float bf2f(unsigned short h) {
  return __uint_as_float(((unsigned)h) << 16);
}

// ---- cost matrix entry -----------------------------------------------------
__device__ __forceinline__ float cost_ij(int i, int j,
                                         const float* ab, const float* ad,
                                         const float* bb, const float* bd) {
  if (i < KTOP) {
    if (j < KTOP)
      return fmaxf(fabsf(ab[i] - bb[j]), fabsf(ad[i] - bd[j]));
    return 0.5f * (ad[i] - ab[i]);
  }
  if (j < KTOP) return 0.5f * (bd[j] - bb[j]);
  return 0.0f;
}

// ===========================================================================
// 1) pooling: softmax over C=4, drop background, 8x8 mean -> 32x32; same for
//    one-hot labels. One thread per (img, pooled cell).
// ===========================================================================
__global__ __launch_bounds__(256)
void pool_kernel(const float* __restrict__ x, const int* __restrict__ y,
                 float* __restrict__ mx, float* __restrict__ my) {
  int t = blockIdx.x * blockDim.x + threadIdx.x;
  if (t >= 12 * NPIX) return;
  int img = t >> 10, cell = t & 1023;
  int b = img / 3, ch = img % 3 + 1;
  int gy = cell >> 5, gx = cell & 31;
  float sx = 0.f, sy = 0.f;
  for (int ry = 0; ry < 8; ++ry) {
    int row = gy * 8 + ry;
    for (int rx = 0; rx < 8; ++rx) {
      int col = gx * 8 + rx;
      int pix = row * 256 + col;
      const float* xb = x + (size_t)b * 4 * 65536 + pix;
      float x0 = xb[0], x1 = xb[65536], x2 = xb[131072], x3 = xb[196608];
      float m = fmaxf(fmaxf(x0, x1), fmaxf(x2, x3));
      float e0 = __expf(x0 - m), e1 = __expf(x1 - m);
      float e2 = __expf(x2 - m), e3 = __expf(x3 - m);
      float inv = 1.f / (e0 + e1 + e2 + e3);
      float ec = (ch == 1) ? e1 : (ch == 2) ? e2 : e3;
      sx += ec * inv;
      int yv = y[(size_t)b * 65536 + pix];
      sy += (yv == ch) ? 1.f : 0.f;
    }
  }
  mx[t] = sx * (1.f / 64.f);
  my[t] = sy * (1.f / 64.f);
}

// ===========================================================================
// 2) persistence diagrams. Block = one (src, sign, img) run.
//    r = src*24 + sign*12 + img ; problem p = sign*12 + img ;
//    diag base = diag + p*1024 + src*512 (ab,ad for src=0; bb,bd for src=1).
// ===========================================================================
__global__ __launch_bounds__(1024)
void diagram_kernel(const float* __restrict__ mx, const float* __restrict__ my,
                    float* __restrict__ diag) {
  __shared__ float    v[NPIX];
  __shared__ float    ekey[SORTN];
  __shared__ unsigned epay[SORTN];
  __shared__ int      parent[NPIX];
  __shared__ float    birth[NPIX];
  __shared__ float    pers[SORTN];
  __shared__ float    pbv[SORTN];
  __shared__ float    pdv[SORTN];

  int r    = blockIdx.x;
  int img  = r % 12;
  int sign = (r / 12) & 1;
  int src  = r / 24;
  int tid  = threadIdx.x;

  const float* vin = (src ? my : mx) + img * NPIX;
  v[tid] = sign ? -vin[tid] : vin[tid];
  parent[tid] = tid;
  __syncthreads();
  birth[tid] = v[tid];

  // edges: e<992 horizontal (u=r*32+c, c<31, v=u+1), else vertical (v=u+32)
  for (int e = tid; e < SORTN; e += blockDim.x) {
    if (e < NE) {
      int u, w2;
      if (e < 992) { int rr = e / 31, cc = e % 31; u = rr * 32 + cc; w2 = u + 1; }
      else         { int e2 = e - 992; int rr = e2 >> 5, cc = e2 & 31; u = rr * 32 + cc; w2 = u + 32; }
      ekey[e] = fmaxf(v[u], v[w2]);
      epay[e] = (unsigned)(u * 1024 + w2);
    } else { ekey[e] = 1e30f; epay[e] = 0u; }
    pers[e] = -1e30f; pbv[e] = 0.f; pdv[e] = 0.f;
  }
  __syncthreads();

  // bitonic sort ascending by ekey (2048 slots, 1024 threads)
  for (int k = 2; k <= SORTN; k <<= 1)
    for (int j = k >> 1; j > 0; j >>= 1) {
      __syncthreads();
      for (int t = tid; t < SORTN; t += blockDim.x) {
        int ixj = t ^ j;
        if (ixj > t) {
          bool up = ((t & k) == 0);
          float a = ekey[t], b = ekey[ixj];
          if ((a > b) == up) {
            ekey[t] = b; ekey[ixj] = a;
            unsigned pa = epay[t]; epay[t] = epay[ixj]; epay[ixj] = pa;
          }
        }
      }
    }
  __syncthreads();

  // serial Kruskal with elder rule (single lane; 1984 iterations)
  if (tid == 0) {
    for (int e = 0; e < NE; ++e) {
      float wv = ekey[e];
      unsigned pay = epay[e];
      int a = (int)(pay >> 10), b = (int)(pay & 1023);
      int ru = a; while (parent[ru] != ru) { parent[ru] = parent[parent[ru]]; ru = parent[ru]; }
      int rv = b; while (parent[rv] != rv) { parent[rv] = parent[parent[rv]]; rv = parent[rv]; }
      float bpt, dpt;
      if (ru != rv) {
        float bu = birth[ru], bv = birth[rv];
        int older   = (bu <= bv) ? ru : rv;
        int younger = (bu <= bv) ? rv : ru;
        parent[younger] = older;
        birth[older] = fminf(bu, bv);
        bpt = fmaxf(bu, bv); dpt = wv;
      } else { bpt = wv; dpt = wv; }
      if (sign) { pers[e] = dpt - bpt; pbv[e] = -dpt; pdv[e] = -bpt; }  // H1 duality
      else      { pers[e] = dpt - bpt; pbv[e] = bpt;  pdv[e] = dpt;  }
    }
    if (!sign) {  // essential H0 class (min, max)
      float mn = v[0], mxv = v[0];
      for (int i = 1; i < NPIX; ++i) { mn = fminf(mn, v[i]); mxv = fmaxf(mxv, v[i]); }
      pers[NE] = mxv - mn; pbv[NE] = mn; pdv[NE] = mxv;
    }
  }
  __syncthreads();

  // bitonic sort DESCENDING by persistence (keep top-256)
  for (int k = 2; k <= SORTN; k <<= 1)
    for (int j = k >> 1; j > 0; j >>= 1) {
      __syncthreads();
      for (int t = tid; t < SORTN; t += blockDim.x) {
        int ixj = t ^ j;
        if (ixj > t) {
          bool up = ((t & k) == 0);
          float a = pers[t], b = pers[ixj];
          if ((a < b) == up) {
            pers[t] = b; pers[ixj] = a;
            float x = pbv[t]; pbv[t] = pbv[ixj]; pbv[ixj] = x;
            x = pdv[t]; pdv[t] = pdv[ixj]; pdv[ixj] = x;
          }
        }
      }
    }
  __syncthreads();

  int p = sign * 12 + img;
  float* base = diag + p * 1024 + src * 512;
  if (tid < KTOP) { base[tid] = pbv[tid]; base[KTOP + tid] = pdv[tid]; }
}

// ===========================================================================
// 3) eps_p = 0.02 * max(max(C), 1e-6)
// ===========================================================================
__global__ __launch_bounds__(256)
void eps_kernel(const float* __restrict__ diag, float* __restrict__ epsArr) {
  __shared__ float red[256];
  int p = blockIdx.x, tid = threadIdx.x;
  const float* ab = diag + p * 1024;
  const float* ad = ab + 256; const float* bb = ad + 256; const float* bd = bb + 256;
  float m = 0.f;
  for (int idx = tid; idx < TWOK * TWOK; idx += 256) {
    int i = idx >> 9, j = idx & 511;
    m = fmaxf(m, cost_ij(i, j, ab, ad, bb, bd));
  }
  red[tid] = m; __syncthreads();
  for (int s = 128; s > 0; s >>= 1) {
    if (tid < s) red[tid] = fmaxf(red[tid], red[tid + s]);
    __syncthreads();
  }
  if (tid == 0) epsArr[p] = 0.02f * fmaxf(red[0], 1e-6f);
}

// ===========================================================================
// 4) K = exp(-C/eps) bf16, row-major (Kf) and transposed (Kt)
// ===========================================================================
__global__ __launch_bounds__(256)
void buildK_kernel(const float* __restrict__ diag, const float* __restrict__ epsArr,
                   unsigned short* __restrict__ Kf, unsigned short* __restrict__ Kt) {
  int blk = blockIdx.x;          // 24*512 blocks
  int p = blk >> 9, i = blk & 511;
  int tid = threadIdx.x;
  const float* ab = diag + p * 1024;
  const float* ad = ab + 256; const float* bb = ad + 256; const float* bd = bb + 256;
  float inv_eps = 1.0f / epsArr[p];
  unsigned short* kf = Kf + (size_t)p * TWOK * TWOK;
  unsigned short* kt = Kt + (size_t)p * TWOK * TWOK;
  for (int j = tid; j < TWOK; j += 256) {
    float c = cost_ij(i, j, ab, ad, bb, bd);
    unsigned short kb = f2bf(__expf(-c * inv_eps));
    kf[i * TWOK + j] = kb;
    kt[j * TWOK + i] = kb;
  }
}

// ===========================================================================
// 5) Sinkhorn via WMMA bf16 matvec.  One 256-thread (8 wave32) block per
//    problem.  matvec: D(16x16) = A(16x32 K-tile) x B(32x16 replicated vec),
//    f32 accumulate over 16 k-tiles; 32 row tiles split 4-per-wave.
//    __noinline__ keeps LLVM from hoisting the (loop-invariant) K loads out
//    of the 150-iteration loop and spilling them to scratch; K stays in L2.
//    Explicit address spaces: K via global_load_b128, vector/output via ds.
// ===========================================================================
__device__ __noinline__ void matvec_wmma(const unsigned short* Kmat,
                                         const unsigned short* vbf_lds,
                                         float* out_lds, int tid) {
  const int wave = tid >> 5;
  const int lane = tid & 31;
  const int half = lane >> 4;      // 0: lanes 0-15, 1: lanes 16-31
  const int mrow = lane & 15;
  GQ4 K4 = (GQ4)(uintptr_t)Kmat;                 // global AS
  LQ4 V4 = (LQ4)(uintptr_t)vbf_lds;              // LDS AS (32-bit offset)
  LF  O  = (LF)(uintptr_t)out_lds;
  const v8f zero = {0.f, 0.f, 0.f, 0.f, 0.f, 0.f, 0.f, 0.f};
  v8f acc[4];
#pragma unroll
  for (int t = 0; t < 4; ++t) acc[t] = zero;

#pragma unroll
  for (int kt = 0; kt < 16; ++kt) {
    int k0 = kt * 32;
    // B tile: vector replicated over all 16 columns.
    // lanes 0-15 hold K=k0..k0+7 & k0+16..k0+23, lanes 16-31: +8 on both.
    int kb = k0 + 8 * half;
    U4x2 breg;
    breg.lo = V4[kb >> 3];
    breg.hi = V4[(kb >> 3) + 2];
    v16bf bv = __builtin_bit_cast(v16bf, breg);
    // A tiles: row-major K.  lanes 0-15: K=k0..k0+7 & k0+16..k0+23;
    //                        lanes 16-31: +8 on both (ISA 16-bit A layout).
    int koff = k0 + half * 8;
#pragma unroll
    for (int t = 0; t < 4; ++t) {
      int row = (wave * 4 + t) * 16 + mrow;
      U4x2 areg;
      areg.lo = K4[(row * TWOK + koff) >> 3];
      areg.hi = K4[((row * TWOK + koff) >> 3) + 2];
      v16bf av = __builtin_bit_cast(v16bf, areg);
      acc[t] = __builtin_amdgcn_wmma_f32_16x16x32_bf16(
          /*neg_a=*/false, av, /*neg_b=*/false, bv,
          /*c_mod=*/(short)0, acc[t], /*reuse_a=*/false, /*reuse_b=*/false);
    }
  }

  // D layout: lanes 0-15 VGPR r -> M=r ; lanes 16-31 VGPR r -> M=r+8.
#pragma unroll
  for (int t = 0; t < 4; ++t) {
    LF o = O + (wave * 4 + t) * 16 + half * 8;
#pragma unroll
    for (int rr = 0; rr < 8; ++rr) o[rr] = acc[t][rr];  // 16 lanes dup-write same value
  }
}

__global__ __launch_bounds__(256)
void sinkhorn_kernel(const float* __restrict__ diag, const float* __restrict__ epsArr,
                     const unsigned short* __restrict__ Kf,
                     const unsigned short* __restrict__ Kt,
                     float* __restrict__ partial) {
  __shared__ __align__(16) float u[TWOK];
  __shared__ __align__(16) float vv[TWOK];
  __shared__ __align__(16) float tmp[TWOK];
  __shared__ __align__(16) unsigned short ubf[TWOK];
  __shared__ __align__(16) unsigned short vbf[TWOK];
  __shared__ float sab[256], sad[256], sbb[256], sbd[256];
  __shared__ float red[256];

  int p = blockIdx.x, tid = threadIdx.x;
  const float* dg = diag + p * 1024;
  sab[tid] = dg[tid]; sad[tid] = dg[256 + tid];
  sbb[tid] = dg[512 + tid]; sbd[tid] = dg[768 + tid];

  vv[tid] = 1.f; vv[tid + 256] = 1.f;
  unsigned short one = f2bf(1.f);
  vbf[tid] = one; vbf[tid + 256] = one;
  __syncthreads();

  const unsigned short* kf = Kf + (size_t)p * TWOK * TWOK;
  const unsigned short* kt = Kt + (size_t)p * TWOK * TWOK;

#pragma unroll 1
  for (int it = 0; it < SINK_ITERS; ++it) {
    asm volatile("" ::: "memory");           // block LICM of K loads across iters
    matvec_wmma(kf, vbf, tmp, tid);          // tmp = K v
    __syncthreads();
    {
      float a = 1.f / tmp[tid], b = 1.f / tmp[tid + 256];
      u[tid] = a; u[tid + 256] = b;
      ubf[tid] = f2bf(a); ubf[tid + 256] = f2bf(b);
    }
    __syncthreads();
    matvec_wmma(kt, ubf, tmp, tid);          // tmp = K^T u
    __syncthreads();
    {
      float a = 1.f / tmp[tid], b = 1.f / tmp[tid + 256];
      vv[tid] = a; vv[tid + 256] = b;
      vbf[tid] = f2bf(a); vbf[tid + 256] = f2bf(b);
    }
    __syncthreads();
  }

  // sum(P*C) = sum_ij u_i K_ij C_ij v_j
  float s = 0.f;
  for (int idx = tid; idx < TWOK * TWOK; idx += 256) {
    int i = idx >> 9, j = idx & 511;
    float c = cost_ij(i, j, sab, sad, sbb, sbd);
    s += u[i] * bf2f(kf[idx]) * vv[j] * c;
  }
  red[tid] = s; __syncthreads();
  for (int st = 128; st > 0; st >>= 1) {
    if (tid < st) red[tid] += red[tid + st];
    __syncthreads();
  }
  if (tid == 0) partial[p] = red[0];
}

// ===========================================================================
// 6) final fixed-order reduction (deterministic, no atomics)
// ===========================================================================
__global__ void final_kernel(const float* __restrict__ partial, float* __restrict__ out) {
  if (threadIdx.x == 0 && blockIdx.x == 0) {
    float s = 0.f;
    for (int p = 0; p < NPROB; ++p) s += partial[p];
    out[0] = s * 0.25f;   // / B, B = 4
  }
}

// ===========================================================================
extern "C" void kernel_launch(void* const* d_in, const int* in_sizes, int n_in,
                              void* d_out, int out_size, void* d_ws, size_t ws_size,
                              hipStream_t stream) {
  const float* x = (const float*)d_in[0];   // [4,4,256,256] f32
  const int*   y = (const int*)d_in[1];     // [4,1,256,256] i32
  float* out = (float*)d_out;

  char* ws = (char*)d_ws;
  float* mx      = (float*)(ws + 0);
  float* my      = (float*)(ws + 49152);
  float* diag    = (float*)(ws + 98304);
  float* epsArr  = (float*)(ws + 196608);
  float* partial = (float*)(ws + 196864);
  unsigned short* Kf = (unsigned short*)(ws + 262144);
  unsigned short* Kt = (unsigned short*)(ws + 262144 + 12582912);
  // total workspace needed: 262144 + 2*12582912 = 25,427,968 bytes

  pool_kernel    <<<48,        256, 0, stream>>>(x, y, mx, my);
  diagram_kernel <<<48,       1024, 0, stream>>>(mx, my, diag);
  eps_kernel     <<<NPROB,     256, 0, stream>>>(diag, epsArr);
  buildK_kernel  <<<NPROB*512, 256, 0, stream>>>(diag, epsArr, Kf, Kt);
  sinkhorn_kernel<<<NPROB,     256, 0, stream>>>(diag, epsArr, Kf, Kt, partial);
  final_kernel   <<<1,          32, 0, stream>>>(partial, out);
}
